// GCN_352187318670
// MI455X (gfx1250) — matrix operand order repaired and checked by reading, not compiled, not run on hardware
//
#include <hip/hip_runtime.h>
#include <hip/hip_bf16.h>

typedef __attribute__((ext_vector_type(16))) _Float16 v16h;
typedef __attribute__((ext_vector_type(8)))  float    v8f;

#define NFEAT   128
#define NHID    64
#define NCLS    40
#define NCLSP   48   // padded to 3 tiles of 16

// ---------------- weight conversion (to transposed f16) ----------------
__global__ void gcn_cvt_w1(const float* __restrict__ W1, _Float16* __restrict__ W1hT) {
    int idx = blockIdx.x * blockDim.x + threadIdx.x;          // 64*128
    if (idx >= NHID * NFEAT) return;
    int c = idx / NFEAT, k = idx % NFEAT;
    W1hT[c * NFEAT + k] = (_Float16)W1[k * NHID + c];
}

__global__ void gcn_cvt_w2(const float* __restrict__ W2, _Float16* __restrict__ W2hT) {
    int idx = blockIdx.x * blockDim.x + threadIdx.x;          // 48*64
    if (idx >= NCLSP * NHID) return;
    int c = idx / NHID, k = idx % NHID;
    W2hT[c * NHID + k] = (c < NCLS) ? (_Float16)W2[k * NCLS + c] : (_Float16)0.f;
}

// ---------------- degree / normalization ----------------
__global__ void gcn_deg_init(float* __restrict__ deg, int N) {
    int i = blockIdx.x * blockDim.x + threadIdx.x;
    if (i < N) deg[i] = 1.0f;                                  // self loop
}

__global__ void gcn_deg_edge(const int* __restrict__ ecol, float* __restrict__ deg, int E) {
    int e = blockIdx.x * blockDim.x + threadIdx.x;
    if (e < E) atomicAdd(&deg[ecol[e]], 1.0f);
}

__global__ void gcn_dinv(const float* __restrict__ deg, float* __restrict__ dinv, int N) {
    int i = blockIdx.x * blockDim.x + threadIdx.x;
    if (i < N) dinv[i] = rsqrtf(deg[i]);
}

// ---------------- GEMM1: t1 = x @ W1  (f16 WMMA, f32 accum) ----------------
// Block = 4 waves, one 16-row M tile. A fragments staged once in LDS
// (wave w converts K-step w), each wave owns one 16-col N tile.
__global__ __launch_bounds__(128) void gcn_gemm1(const float* __restrict__ x,
                                                 const _Float16* __restrict__ W1hT,
                                                 float* __restrict__ t1, int N) {
    __shared__ v16h As[4][32];                    // [k-step][lane], 4 KB

    const int lane = threadIdx.x & 31;
    const int wave = threadIdx.x >> 5;            // N-tile 0..3, also staged k-step
    const int mt   = blockIdx.x;
    const int g    = lane >> 4;                   // half-wave group
    const int nl   = lane & 15;

    // A row for this lane; clamped (OOB rows only feed store-guarded outputs)
    int row = mt * 16 + nl;
    row = (row < N) ? row : (N - 1);

    // stage A fragment for k-step == wave
    {
        const float* ap = x + (size_t)row * NFEAT + wave * 32 + g * 16;
        v8f lo = *(const v8f*)ap;                 // K = ks*32 + g*16 .. +7
        v8f hi = *(const v8f*)(ap + 8);           // K = ks*32 + g*16 + 8 .. +15
        v16h a;
#pragma unroll
        for (int i = 0; i < 8; ++i) { a[i] = (_Float16)lo[i]; a[i + 8] = (_Float16)hi[i]; }
        As[wave][lane] = a;
    }
    __syncthreads();

    const int col = wave * 16 + nl;
    v8f c = {};
#pragma unroll
    for (int ks = 0; ks < 4; ++ks) {
        v16h a = As[ks][lane];
        v16h b = *(const v16h*)(W1hT + (size_t)col * NFEAT + ks * 32 + g * 16);
        c = __builtin_amdgcn_wmma_f32_16x16x32_f16(false, a, false, b, (short)0, c, false, false);
    }

    // store: single base pointer + compile-time-constant strides (immediate offsets)
    const int m0 = mt * 16 + g * 8;               // first of this half-wave's 8 rows
    float* op = t1 + (size_t)m0 * NHID + wave * 16 + nl;
    if (m0 + 8 <= N) {                            // fast path: whole half-tile in range
#pragma unroll
        for (int r = 0; r < 8; ++r) op[r * NHID] = c[r];
    } else {
#pragma unroll
        for (int r = 0; r < 8; ++r) if (m0 + r < N) op[r * NHID] = c[r];
    }
}

// ---------------- agg1 init: agg[i] = b1 + t1[i]*dinv[i]^2 (self loop + bias) ----------------
__global__ void gcn_agg1_init(const float* __restrict__ t1, const float* __restrict__ dinv,
                              const float* __restrict__ b1, float* __restrict__ agg, int N) {
    long long idx = (long long)blockIdx.x * blockDim.x + threadIdx.x;
    if (idx >= (long long)N * NHID) return;
    int i = (int)(idx >> 6), j = (int)(idx & 63);
    float di = dinv[i];
    agg[idx] = b1[j] + t1[idx] * di * di;
}

// ---------------- agg1 edge scatter: agg[col] += t1[row] * dinv[row]*dinv[col] ----------------
__global__ void gcn_agg1_edge(const int* __restrict__ er, const int* __restrict__ ec,
                              const float* __restrict__ dinv, const float* __restrict__ t1,
                              float* __restrict__ agg, int E) {
    long long t = (long long)blockIdx.x * blockDim.x + threadIdx.x;
    int e = (int)(t >> 4);                        // 16 chunks of 4 features
    if (e >= E) return;
    int j = (int)(t & 15) * 4;
    int r = er[e], c = ec[e];
    float w = dinv[r] * dinv[c];
    float4 v = *(const float4*)(t1 + (size_t)r * NHID + j);
    float* o = agg + (size_t)c * NHID + j;
    atomicAdd(o + 0, v.x * w);
    atomicAdd(o + 1, v.y * w);
    atomicAdd(o + 2, v.z * w);
    atomicAdd(o + 3, v.w * w);
}

// ---------------- GEMM2: t2 = relu(agg1) @ W2  (relu folded into A stage) ----------------
__global__ __launch_bounds__(96) void gcn_gemm2(const float* __restrict__ h,
                                                const _Float16* __restrict__ W2hT,
                                                float* __restrict__ t2, int N) {
    __shared__ v16h As[2][32];                    // [k-step][lane], 2 KB

    const int lane = threadIdx.x & 31;
    const int wave = threadIdx.x >> 5;            // N-tile 0..2
    const int mt   = blockIdx.x;
    const int g    = lane >> 4;
    const int nl   = lane & 15;

    int row = mt * 16 + nl;
    row = (row < N) ? row : (N - 1);

    if (wave < 2) {                               // waves 0,1 stage the 2 k-steps
        const float* ap = h + (size_t)row * NHID + wave * 32 + g * 16;
        v8f lo = *(const v8f*)ap;
        v8f hi = *(const v8f*)(ap + 8);
        v16h a;
#pragma unroll
        for (int i = 0; i < 8; ++i) {
            a[i]     = (_Float16)fmaxf(lo[i], 0.0f);          // ReLU on load
            a[i + 8] = (_Float16)fmaxf(hi[i], 0.0f);
        }
        As[wave][lane] = a;
    }
    __syncthreads();

    const int col = wave * 16 + nl;
    v8f c = {};
#pragma unroll
    for (int ks = 0; ks < 2; ++ks) {
        v16h a = As[ks][lane];
        v16h b = *(const v16h*)(W2hT + (size_t)col * NHID + ks * 32 + g * 16);
        c = __builtin_amdgcn_wmma_f32_16x16x32_f16(false, a, false, b, (short)0, c, false, false);
    }

    const int m0 = mt * 16 + g * 8;
    const int n  = wave * 16 + nl;
    float* op = t2 + (size_t)m0 * NCLS + n;
    if (n < NCLS) {                               // lane predicate applied once
        if (m0 + 8 <= N) {
#pragma unroll
            for (int r = 0; r < 8; ++r) op[r * NCLS] = c[r];
        } else {
#pragma unroll
            for (int r = 0; r < 8; ++r) if (m0 + r < N) op[r * NCLS] = c[r];
        }
    }
}

// ---------------- out init: out[i] = b2 + t2[i]*dinv[i]^2 ----------------
__global__ void gcn_out_init(const float* __restrict__ t2, const float* __restrict__ dinv,
                             const float* __restrict__ b2, float* __restrict__ out, int N) {
    long long idx = (long long)blockIdx.x * blockDim.x + threadIdx.x;
    if (idx >= (long long)N * NCLS) return;
    int i = (int)(idx / NCLS), j = (int)(idx % NCLS);
    float di = dinv[i];
    out[idx] = b2[j] + t2[idx] * di * di;
}

// ---------------- agg2 edge scatter into d_out ----------------
__global__ void gcn_agg2_edge(const int* __restrict__ er, const int* __restrict__ ec,
                              const float* __restrict__ dinv, const float* __restrict__ t2,
                              float* __restrict__ out, int E) {
    long long t = (long long)blockIdx.x * blockDim.x + threadIdx.x;
    int e = (int)(t / 10);                        // 10 chunks of 4 features (40 total)
    if (e >= E) return;
    int j = (int)(t % 10) * 4;
    int r = er[e], c = ec[e];
    float w = dinv[r] * dinv[c];
    float4 v = *(const float4*)(t2 + (size_t)r * NCLS + j);
    float* o = out + (size_t)c * NCLS + j;
    atomicAdd(o + 0, v.x * w);
    atomicAdd(o + 1, v.y * w);
    atomicAdd(o + 2, v.z * w);
    atomicAdd(o + 3, v.w * w);
}

extern "C" void kernel_launch(void* const* d_in, const int* in_sizes, int n_in,
                              void* d_out, int out_size, void* d_ws, size_t ws_size,
                              hipStream_t stream) {
    const float* x     = (const float*)d_in[0];
    const int*   edges = (const int*)d_in[1];
    const float* W1    = (const float*)d_in[2];
    const float* b1    = (const float*)d_in[3];
    const float* W2    = (const float*)d_in[4];
    const float* b2    = (const float*)d_in[5];
    float*       out   = (float*)d_out;

    const int N = in_sizes[0] / NFEAT;            // 100000
    const int E = in_sizes[1] / 2;                // 1200000
    const int* erow = edges;
    const int* ecol = edges + E;

    // workspace layout (floats)
    float*    bufT  = (float*)d_ws;               // N*64 : t1, then reused for t2 (N*40)
    float*    bufA  = bufT + (size_t)N * NHID;    // N*64 : agg1 (= h pre-relu)
    float*    deg   = bufA + (size_t)N * NHID;    // N
    float*    dinv  = deg + N;                    // N
    _Float16* W1hT  = (_Float16*)(dinv + N);      // 64*128
    _Float16* W2hT  = W1hT + NHID * NFEAT;        // 48*64

    const int TB = 256;
    // weights -> transposed f16
    gcn_cvt_w1<<<(NHID * NFEAT + TB - 1) / TB, TB, 0, stream>>>(W1, W1hT);
    gcn_cvt_w2<<<(NCLSP * NHID + TB - 1) / TB, TB, 0, stream>>>(W2, W2hT);

    // degrees + normalization
    gcn_deg_init<<<(N + TB - 1) / TB, TB, 0, stream>>>(deg, N);
    gcn_deg_edge<<<(E + TB - 1) / TB, TB, 0, stream>>>(ecol, deg, E);
    gcn_dinv<<<(N + TB - 1) / TB, TB, 0, stream>>>(deg, dinv, N);

    const int MT = (N + 15) / 16;                 // 6250 row tiles

    // layer 1
    gcn_gemm1<<<MT, 128, 0, stream>>>(x, W1hT, bufT, N);
    {
        long long total = (long long)N * NHID;
        gcn_agg1_init<<<(unsigned)((total + TB - 1) / TB), TB, 0, stream>>>(bufT, dinv, b1, bufA, N);
        long long work = (long long)E * 16;
        gcn_agg1_edge<<<(unsigned)((work + TB - 1) / TB), TB, 0, stream>>>(erow, ecol, dinv, bufT, bufA, E);
    }

    // layer 2 (t2 reuses bufT; relu folded into gemm2 A-stage)
    gcn_gemm2<<<MT, 96, 0, stream>>>(bufA, W2hT, bufT, N);
    {
        long long total = (long long)N * NCLS;
        gcn_out_init<<<(unsigned)((total + TB - 1) / TB), TB, 0, stream>>>(bufT, dinv, b2, out, N);
        long long work = (long long)E * 10;
        gcn_agg2_edge<<<(unsigned)((work + TB - 1) / TB), TB, 0, stream>>>(erow, ecol, dinv, bufT, out, E);
    }
}
